// Generator_36842229465594
// MI455X (gfx1250) — matrix operand order repaired and checked by reading, not compiled, not run on hardware
//
#include <hip/hip_runtime.h>
#include <hip/hip_bf16.h>

typedef __attribute__((ext_vector_type(16))) _Float16 v16h;
typedef __attribute__((ext_vector_type(8)))  _Float16 v8h;
typedef __attribute__((ext_vector_type(8)))  float    v8f;

#define BB 4096
#define TT 16
#define HH 256
#define FF 256
#define ZZ 128
#define PP 10
#define PREV_ 5

// Persistent f16 activation matrix X: row-major B x XS, fixed column regions.
#define XHI    0      // h_i        (256)
#define XHRAW  256    // h_raw      (256)
#define XY     512    // y          (256)
#define XYPREV 768    // y_prev     (256)
#define XZ     1024   // z          (128)
#define XH     1152   // h (lstm)   (256)
#define XS     1408

struct GemmSegs { int aoff[4]; int woff[4]; int len[4]; int nseg; };

// ---------------------------------------------------------------- pack: f32 KxN -> f16 transposed NxK (at column offset)
__global__ void pack_wt_kernel(const float* __restrict__ W, int K, int N,
                               _Float16* __restrict__ WT, int ldw, int coff) {
  int idx = blockIdx.x * 256 + threadIdx.x;
  if (idx >= K * N) return;
  int k = idx / N, n = idx - k * N;
  WT[(size_t)n * ldw + coff + k] = (_Float16)W[idx];
}

// ---------------------------------------------------------------- init recurrent state
__global__ void init_state_kernel(const float* __restrict__ h_i,
                                  _Float16* __restrict__ X,
                                  float* __restrict__ c) {
  int idx = blockIdx.x * 256 + threadIdx.x;  // B*256 threads exactly
  int b = idx >> 8, n = idx & 255;
  _Float16* xr = X + (size_t)b * XS;
  xr[XHI + n]    = (_Float16)h_i[(size_t)b * HH + n];
  xr[XYPREV + n] = (_Float16)0.0f;
  xr[XH + n]     = (_Float16)0.0f;
  if (n < ZZ) xr[XZ + n] = (_Float16)0.0f;
  c[idx] = 0.0f;
}

// ---------------------------------------------------------------- Hawkes intensity: lam, prob
__global__ void hawkes_kernel(const float* __restrict__ input_t,
                              const float* __restrict__ alpha_p,
                              const float* __restrict__ beta_p,
                              const float* __restrict__ base_p,
                              float* __restrict__ lam_out,
                              float* __restrict__ probs_out,
                              int cti) {
  int b = blockIdx.x * 256 + threadIdx.x;
  if (b >= BB) return;
  const float alpha = alpha_p[0], beta = beta_p[0], base = base_p[0];
  const float* row = input_t + (size_t)b * TT;
  float ct = row[cti];
  float last = row[cti - 1];
  float tk = 0.f, td3 = 0.f;
  for (int i = 0; i < cti; ++i) {
    float t = row[i];
    tk  += __expf(beta * (t - ct));
    td3 += __expf(beta * (t - last));
  }
  float lam = base + alpha * tk;
  float prob = lam * __expf((last - ct) * base + (alpha / beta) * (tk - td3));
  lam_out[b] = lam;
  probs_out[b] = prob;
}

// ---------------------------------------------------------------- WMMA fragment load helpers
// A (16x32 f16, MxK): lane L holds row M=L&15; halves 0..15 = K k0..k0+15, k0=(L>>4)*16.
__device__ __forceinline__ v16h load_a_frag(const _Float16* p) {
  v8h lo = *(const v8h*)p;          // K  +0.. +7
  v8h hi = *(const v8h*)(p + 8);    // K  +8..+15
  v16h r;
#pragma unroll
  for (int t = 0; t < 8; ++t) { r[t] = lo[t]; r[t + 8] = hi[t]; }
  return r;
}

// ---------------------------------------------------------------- fused segmented GEMM: C = act(sum_s A[:,aoff+k] * WT[n, woff+k] + bias)
// A: f16 row-major (lda), WT: f16 N x Ktot (pre-transposed weights, ldw), M = BB.
// Workgroup = 4 waves = 64x64 C tile; each wave = 32x32 via 2x2 WMMA accumulators.
__global__ void __launch_bounds__(128)
gemm_kernel(const _Float16* __restrict__ A, int lda,
            const _Float16* __restrict__ WT, int ldw,
            const float* __restrict__ bias,
            float* __restrict__ outF32, long long osF32,
            _Float16* __restrict__ outF16, long long osF16,
            int relu, GemmSegs segs) {
  const int lane  = threadIdx.x & 31;
  const int wave  = threadIdx.x >> 5;
  const int mrow0 = blockIdx.x * 64 + (wave & 1) * 32;
  const int ncol0 = blockIdx.y * 64 + (wave >> 1) * 32;
  const int l15   = lane & 15;
  const int khalf = lane >> 4;

  v8f zero = {};
  v8f acc[2][2];
#pragma unroll
  for (int i = 0; i < 2; ++i)
#pragma unroll
    for (int jn = 0; jn < 2; ++jn) acc[i][jn] = zero;

  for (int s = 0; s < segs.nseg; ++s) {
    const int aoff = segs.aoff[s], woff = segs.woff[s], len = segs.len[s];
    const _Float16* A0 = A  + (size_t)(mrow0      + l15) * lda + aoff + khalf * 16;
    const _Float16* A1 = A  + (size_t)(mrow0 + 16 + l15) * lda + aoff + khalf * 16;
    // B (32x16 f16, KxN): lane holds col N=L&15; halves = contiguous K block, base (L>>4)*16.
    const _Float16* B0 = WT + (size_t)(ncol0      + l15) * ldw + woff + khalf * 16;
    const _Float16* B1 = WT + (size_t)(ncol0 + 16 + l15) * ldw + woff + khalf * 16;
    for (int kk = 0; kk < len; kk += 32) {
      v16h a0 = load_a_frag(A0 + kk);
      v16h a1 = load_a_frag(A1 + kk);
      v16h b0 = *(const v16h*)(B0 + kk);
      v16h b1 = *(const v16h*)(B1 + kk);
      acc[0][0] = __builtin_amdgcn_wmma_f32_16x16x32_f16(false, a0, false, b0, (short)0, acc[0][0], false, false);
      acc[0][1] = __builtin_amdgcn_wmma_f32_16x16x32_f16(false, a0, false, b1, (short)0, acc[0][1], false, false);
      acc[1][0] = __builtin_amdgcn_wmma_f32_16x16x32_f16(false, a1, false, b0, (short)0, acc[1][0], false, false);
      acc[1][1] = __builtin_amdgcn_wmma_f32_16x16x32_f16(false, a1, false, b1, (short)0, acc[1][1], false, false);
    }
  }

  // C/D f32 16x16 layout: VGPR r, lanes 0-15 -> M=r, lanes 16-31 -> M=r+8; N = lane&15.
#pragma unroll
  for (int i = 0; i < 2; ++i) {
#pragma unroll
    for (int jn = 0; jn < 2; ++jn) {
      int col = ncol0 + jn * 16 + l15;
      float bv = bias ? bias[col] : 0.0f;
#pragma unroll
      for (int r = 0; r < 8; ++r) {
        int row = mrow0 + i * 16 + khalf * 8 + r;
        float v = acc[i][jn][r] + bv;
        if (relu) v = v > 0.0f ? v : 0.0f;
        if (outF32) outF32[(size_t)row * (size_t)osF32 + col] = v;
        if (outF16) outF16[(size_t)row * (size_t)osF16 + col] = (_Float16)v;
      }
    }
  }
}

// ---------------------------------------------------------------- LSTM cell epilogue
__device__ __forceinline__ float sigf(float x) { return 1.0f / (1.0f + __expf(-x)); }

__global__ void lstm_cell_kernel(const float* __restrict__ gates,
                                 float* __restrict__ c,
                                 const float* __restrict__ lam,
                                 _Float16* __restrict__ X) {
  int idx = blockIdx.x * 256 + threadIdx.x;   // B*256 exactly
  int b = idx >> 8, n = idx & 255;
  const float* g = gates + (size_t)b * 1024;
  float gi = g[n], gf = g[256 + n], gg = g[512 + n], go = g[768 + n];
  float cn = sigf(gf) * c[idx] + sigf(gi) * tanhf(gg);
  float hraw = sigf(go) * tanhf(cn);
  c[idx] = cn;
  _Float16* xr = X + (size_t)b * XS;
  xr[XHRAW + n] = (_Float16)hraw;
  xr[XH + n]    = (_Float16)(lam[b] * hraw);
}

// ---------------------------------------------------------------- y_prev <- y
__global__ void advance_kernel(_Float16* __restrict__ X) {
  int idx = blockIdx.x * 256 + threadIdx.x;
  int b = idx >> 8, n = idx & 255;
  _Float16* xr = X + (size_t)b * XS;
  xr[XYPREV + n] = xr[XY + n];
}

// ---------------------------------------------------------------- reparameterization: z = m + eps*sqrt(exp(lv))
__global__ void z_kernel(const float* __restrict__ m_post_sl,   // + j*Z, row stride P*Z
                         const float* __restrict__ lv_post_sl,
                         const float* __restrict__ m_prior_sl,
                         const float* __restrict__ lv_prior_sl,
                         const float* __restrict__ noise_post_j,  // + j*B*Z, dense
                         const float* __restrict__ noise_prior_j,
                         float* __restrict__ z_post_j,
                         float* __restrict__ z_prior_j,
                         _Float16* __restrict__ X) {
  int idx = blockIdx.x * 256 + threadIdx.x;  // B*Z exactly
  int b = idx >> 7, n = idx & 127;
  size_t rp = (size_t)b * (PP * ZZ) + n;
  float zp = m_post_sl[rp] + noise_post_j[idx] * sqrtf(__expf(lv_post_sl[rp]));
  z_post_j[idx]  = zp;
  z_prior_j[idx] = m_prior_sl[rp] + noise_prior_j[idx] * sqrtf(__expf(lv_prior_sl[rp]));
  X[(size_t)b * XS + XZ + n] = (_Float16)zp;
}

// ================================================================ host
extern "C" void kernel_launch(void* const* d_in, const int* in_sizes, int n_in,
                              void* d_out, int out_size, void* d_ws, size_t ws_size,
                              hipStream_t stream) {
  const float* h_i     = (const float*)d_in[0];
  const float* input_t = (const float*)d_in[1];
  const float* noise_post  = (const float*)d_in[2];
  const float* noise_prior = (const float*)d_in[3];
  const float* W_lstm = (const float*)d_in[4];
  const float* U_lstm = (const float*)d_in[5];
  const float* b_lstm = (const float*)d_in[6];
  const float* W1 = (const float*)d_in[7];  const float* b1 = (const float*)d_in[8];
  const float* W2 = (const float*)d_in[9];  const float* b2 = (const float*)d_in[10];
  const float* W3 = (const float*)d_in[11]; const float* b3 = (const float*)d_in[12];
  const float* W4 = (const float*)d_in[13]; const float* b4 = (const float*)d_in[14];
  const float* W5 = (const float*)d_in[15]; const float* b5 = (const float*)d_in[16];
  const float* W6 = (const float*)d_in[17]; const float* b6 = (const float*)d_in[18];
  const float* W7 = (const float*)d_in[19]; const float* b7 = (const float*)d_in[20];
  const float* W8 = (const float*)d_in[21]; const float* b8 = (const float*)d_in[22];
  const float* W9 = (const float*)d_in[23]; const float* b9 = (const float*)d_in[24];
  const float* alpha = (const float*)d_in[25];
  const float* beta  = (const float*)d_in[26];
  const float* base  = (const float*)d_in[27];

  // output slices, flat in return order
  float* out      = (float*)d_out;
  float* ys       = out;
  float* m_post   = ys       + (size_t)BB * PP * FF;
  float* lv_post  = m_post   + (size_t)BB * PP * ZZ;
  float* m_prior  = lv_post  + (size_t)BB * PP * ZZ;
  float* lv_prior = m_prior  + (size_t)BB * PP * ZZ;
  float* z_post   = lv_prior + (size_t)BB * PP * ZZ;
  float* z_prior  = z_post   + (size_t)PP * BB * ZZ;
  float* probs    = z_prior  + (size_t)PP * BB * ZZ;

  // workspace carve-up
  char* ws = (char*)d_ws;
  size_t off = 0;
  auto take = [&](size_t bytes) -> char* {
    char* p = ws + off;
    off = (off + bytes + 255) & ~(size_t)255;
    return p;
  };
  _Float16* WgT  = (_Float16*)take((size_t)1024 * 896 * 2);  // [W_lstm | U_lstm]^T
  _Float16* W1T  = (_Float16*)take((size_t)256 * 256 * 2);
  _Float16* W2T  = (_Float16*)take((size_t)256 * 256 * 2);
  _Float16* W3T  = (_Float16*)take((size_t)256 * 256 * 2);
  _Float16* W4T  = (_Float16*)take((size_t)128 * 1024 * 2);
  _Float16* W5T  = (_Float16*)take((size_t)128 * 128 * 2);
  _Float16* W6T  = (_Float16*)take((size_t)128 * 128 * 2);
  _Float16* W7T  = (_Float16*)take((size_t)128 * 768 * 2);
  _Float16* W8T  = (_Float16*)take((size_t)128 * 128 * 2);
  _Float16* W9T  = (_Float16*)take((size_t)128 * 128 * 2);
  _Float16* Xb   = (_Float16*)take((size_t)BB * XS * 2);
  _Float16* y1t  = (_Float16*)take((size_t)BB * FF * 2);
  _Float16* y2t  = (_Float16*)take((size_t)BB * FF * 2);
  _Float16* hz16 = (_Float16*)take((size_t)BB * ZZ * 2);
  _Float16* hzp16= (_Float16*)take((size_t)BB * ZZ * 2);
  float* gates   = (float*)take((size_t)BB * 1024 * 4);
  float* cbuf    = (float*)take((size_t)BB * HH * 4);
  float* lam     = (float*)take((size_t)BB * 4);

  auto packN = [&](const float* W, int K, int N, _Float16* WT, int ldw, int coff) {
    int tot = K * N;
    pack_wt_kernel<<<(tot + 255) / 256, 256, 0, stream>>>(W, K, N, WT, ldw, coff);
  };
  packN(W_lstm, 640, 1024, WgT, 896, 0);
  packN(U_lstm, 256, 1024, WgT, 896, 640);
  packN(W1, 256, 256, W1T, 256, 0);
  packN(W2, 256, 256, W2T, 256, 0);
  packN(W3, 256, 256, W3T, 256, 0);
  packN(W4, 1024, 128, W4T, 1024, 0);
  packN(W5, 128, 128, W5T, 128, 0);
  packN(W6, 128, 128, W6T, 128, 0);
  packN(W7, 768, 128, W7T, 768, 0);
  packN(W8, 128, 128, W8T, 128, 0);
  packN(W9, 128, 128, W9T, 128, 0);

  init_state_kernel<<<BB, 256, 0, stream>>>(h_i, Xb, cbuf);

  const dim3 blk(128);
  for (int j = 0; j < PP; ++j) {
    int cti = PREV_ + j;
    hawkes_kernel<<<(BB + 255) / 256, 256, 0, stream>>>(
        input_t, alpha, beta, base, lam, probs + (size_t)j * BB, cti);

    // gates = [z, h_i, y_prev] @ W_lstm + h @ U_lstm + b_lstm    (K segments vs fused WgT)
    GemmSegs gg;
    gg.nseg = 4;
    gg.aoff[0] = XZ;     gg.woff[0] = 0;   gg.len[0] = 128;
    gg.aoff[1] = XHI;    gg.woff[1] = 128; gg.len[1] = 256;
    gg.aoff[2] = XYPREV; gg.woff[2] = 384; gg.len[2] = 256;
    gg.aoff[3] = XH;     gg.woff[3] = 640; gg.len[3] = 256;
    gemm_kernel<<<dim3(BB / 64, 1024 / 64), blk, 0, stream>>>(
        Xb, XS, WgT, 896, b_lstm, gates, 1024LL, (_Float16*)nullptr, 0LL, 0, gg);

    lstm_cell_kernel<<<BB, 256, 0, stream>>>(gates, cbuf, lam, Xb);

    GemmSegs sH; sH.nseg = 1; sH.aoff[0] = XHRAW; sH.woff[0] = 0; sH.len[0] = 256;
    gemm_kernel<<<dim3(BB / 64, FF / 64), blk, 0, stream>>>(
        Xb, XS, W1T, 256, b1, (float*)nullptr, 0LL, y1t, (long long)FF, 1, sH);

    GemmSegs s0; s0.nseg = 1; s0.aoff[0] = 0; s0.woff[0] = 0; s0.len[0] = 256;
    gemm_kernel<<<dim3(BB / 64, FF / 64), blk, 0, stream>>>(
        y1t, FF, W2T, 256, b2, (float*)nullptr, 0LL, y2t, (long long)FF, 1, s0);

    // y3 -> ys output slice (B,P,F) and back into X's y region
    gemm_kernel<<<dim3(BB / 64, FF / 64), blk, 0, stream>>>(
        y2t, FF, W3T, 256, b3, ys + (size_t)j * FF, (long long)(PP * FF),
        Xb + XY, (long long)XS, 1, s0);

    // hz = relu([h_i, h_raw, y, y_prev] @ W4 + b4)
    GemmSegs g4;
    g4.nseg = 4;
    g4.aoff[0] = XHI;    g4.woff[0] = 0;   g4.len[0] = 256;
    g4.aoff[1] = XHRAW;  g4.woff[1] = 256; g4.len[1] = 256;
    g4.aoff[2] = XY;     g4.woff[2] = 512; g4.len[2] = 256;
    g4.aoff[3] = XYPREV; g4.woff[3] = 768; g4.len[3] = 256;
    gemm_kernel<<<dim3(BB / 64, ZZ / 64), blk, 0, stream>>>(
        Xb, XS, W4T, 1024, b4, (float*)nullptr, 0LL, hz16, (long long)ZZ, 1, g4);

    // hz_p = relu([h_i, h_raw, y_prev] @ W7 + b7)
    GemmSegs g7;
    g7.nseg = 3;
    g7.aoff[0] = XHI;    g7.woff[0] = 0;   g7.len[0] = 256;
    g7.aoff[1] = XHRAW;  g7.woff[1] = 256; g7.len[1] = 256;
    g7.aoff[2] = XYPREV; g7.woff[2] = 512; g7.len[2] = 256;
    gemm_kernel<<<dim3(BB / 64, ZZ / 64), blk, 0, stream>>>(
        Xb, XS, W7T, 768, b7, (float*)nullptr, 0LL, hzp16, (long long)ZZ, 1, g7);

    advance_kernel<<<BB, 256, 0, stream>>>(Xb);  // y_prev <- y

    GemmSegs sz; sz.nseg = 1; sz.aoff[0] = 0; sz.woff[0] = 0; sz.len[0] = 128;
    gemm_kernel<<<dim3(BB / 64, ZZ / 64), blk, 0, stream>>>(
        hz16, ZZ, W5T, 128, b5, m_post + (size_t)j * ZZ, (long long)(PP * ZZ),
        (_Float16*)nullptr, 0LL, 1, sz);
    gemm_kernel<<<dim3(BB / 64, ZZ / 64), blk, 0, stream>>>(
        hz16, ZZ, W6T, 128, b6, lv_post + (size_t)j * ZZ, (long long)(PP * ZZ),
        (_Float16*)nullptr, 0LL, 1, sz);
    gemm_kernel<<<dim3(BB / 64, ZZ / 64), blk, 0, stream>>>(
        hzp16, ZZ, W8T, 128, b8, m_prior + (size_t)j * ZZ, (long long)(PP * ZZ),
        (_Float16*)nullptr, 0LL, 1, sz);
    gemm_kernel<<<dim3(BB / 64, ZZ / 64), blk, 0, stream>>>(
        hzp16, ZZ, W9T, 128, b9, lv_prior + (size_t)j * ZZ, (long long)(PP * ZZ),
        (_Float16*)nullptr, 0LL, 1, sz);

    z_kernel<<<(BB * ZZ) / 256, 256, 0, stream>>>(
        m_post + (size_t)j * ZZ, lv_post + (size_t)j * ZZ,
        m_prior + (size_t)j * ZZ, lv_prior + (size_t)j * ZZ,
        noise_post + (size_t)j * BB * ZZ, noise_prior + (size_t)j * BB * ZZ,
        z_post + (size_t)j * BB * ZZ, z_prior + (size_t)j * BB * ZZ, Xb);
  }
}
